// TriPlaneGenerator_12060268167234
// MI455X (gfx1250) — compile-verified, matrix operand order checked by reading, not run on hardware
//
#include <hip/hip_runtime.h>
#include <hip/hip_bf16.h>

typedef __attribute__((ext_vector_type(16))) _Float16 v16h;
typedef __attribute__((ext_vector_type(8)))  _Float16 v8h;
typedef __attribute__((ext_vector_type(8)))  float    v8f;

#define C_FEAT   32
#define HIDDEN   64
#define RAYS_R   4096
#define SAMP_S   96
#define NBATCH   2
#define MM       (RAYS_R * SAMP_S)     // 393216
#define TOT      (NBATCH * MM)         // 786432
#define PLANE_HW (256 * 256)

#define LOG2E 1.4426950408889634f
#define LN2   0.6931471805599453f

// exp(x) via raw v_exp_f32 (exp2)
__device__ __forceinline__ float fast_exp(float x) {
  return __builtin_amdgcn_exp2f(x * LOG2E);
}

// Branch-free fast softplus: max(x,0) + log(1+exp(-|x|)).
// exp arg <= 0 so e in (0,1]; log arg in [1,2] -> never denormal, raw v_log_f32 is safe.
__device__ __forceinline__ float softplus_f(float x) {
  float e = __builtin_amdgcn_exp2f(-fabsf(x) * LOG2E);
  return fmaxf(x, 0.0f) + __builtin_amdgcn_logf(1.0f + e) * LN2;
}

// Fast sigmoid: v_exp_f32 + v_rcp_f32.
__device__ __forceinline__ float sigmoid_f(float x) {
  return __builtin_amdgcn_rcpf(1.0f + fast_exp(-x));
}

// ---------------------------------------------------------------------------
// Kernel 1: fused triplane bilinear sampling + 2-layer MLP via WMMA.
// 256 threads = 8 waves; block handles 128 samples (16 per wave).
// ---------------------------------------------------------------------------
__global__ __launch_bounds__(256) void triplane_mlp_kernel(
    const float* __restrict__ planes,   // (N,3,32,256,256)
    const float* __restrict__ coords,   // (N,M,3)
    const float* __restrict__ w1,       // (64,32)
    const float* __restrict__ b1,       // (64)
    const float* __restrict__ w2,       // (33,64)
    const float* __restrict__ b2,       // (33)
    _Float16*    __restrict__ rgb_buf,  // (TOT,32) f16
    float*       __restrict__ sigma_buf)// (TOT)
{
  __shared__ _Float16 x_lds[128][C_FEAT];     // sampled features (f16)
  __shared__ _Float16 w1_lds[HIDDEN][C_FEAT]; // w1 * 1/sqrt(32)
  __shared__ _Float16 w2_lds[48][HIDDEN];     // w2 * 1/sqrt(64), zero-padded rows 33..47
  __shared__ _Float16 h_lds[8][16][72];       // per-wave hidden activations (stride 72 breaks conflicts)

  const int t = threadIdx.x;
  const int block_base = blockIdx.x * 128;

  // ---- phase 0a: stage f16 weights (scales folded) ----
  const float s1 = 0.17677669529663687f; // 1/sqrt(32)
  const float s2 = 0.125f;               // 1/sqrt(64)
  {
    _Float16* w1f = &w1_lds[0][0];
    #pragma unroll
    for (int i = 0; i < 8; ++i) {
      int idx = t * 8 + i;               // 0..2047
      w1f[idx] = (_Float16)(w1[idx] * s1);
    }
    _Float16* w2f = &w2_lds[0][0];
    #pragma unroll
    for (int i = 0; i < 12; ++i) {
      int idx = t * 12 + i;              // 0..3071
      int row = idx >> 6, col = idx & 63;
      float v = (row < 33) ? (w2[row * 64 + col] * s2) : 0.0f;
      w2f[idx] = (_Float16)v;
    }
  }

  // ---- phase 0b: triplane bilinear sampling (2 threads per sample, 16 ch each) ----
  {
    const int s_local = t >> 1;
    const int cg = (t & 1) * 16;
    const int g = block_base + s_local;
    const int n = g / MM;
    const float cx = coords[(size_t)g * 3 + 0];
    const float cy = coords[(size_t)g * 3 + 1];
    const float cz = coords[(size_t)g * 3 + 2];
    const float px[3] = {cx, cx, cz};    // projs: (x0,y1), (x0,z2), (z2,x0)
    const float py[3] = {cy, cz, cx};

    float acc[16];
    #pragma unroll
    for (int i = 0; i < 16; ++i) acc[i] = 0.0f;

    #pragma unroll
    for (int p = 0; p < 3; ++p) {
      float fx = (px[p] + 1.0f) * 128.0f - 0.5f;
      float fy = (py[p] + 1.0f) * 128.0f - 0.5f;
      float x0f = floorf(fx), y0f = floorf(fy);
      float wx1 = fx - x0f, wy1 = fy - y0f;
      float wx0 = 1.0f - wx1, wy0 = 1.0f - wy1;
      int x0 = (int)x0f, y0 = (int)y0f;
      int x1 = x0 + 1, y1 = y0 + 1;
      float vx0 = (x0 >= 0 && x0 < 256) ? 1.0f : 0.0f;
      float vx1 = (x1 >= 0 && x1 < 256) ? 1.0f : 0.0f;
      float vy0 = (y0 >= 0 && y0 < 256) ? 1.0f : 0.0f;
      float vy1 = (y1 >= 0 && y1 < 256) ? 1.0f : 0.0f;
      int x0c = min(max(x0, 0), 255), x1c = min(max(x1, 0), 255);
      int y0c = min(max(y0, 0), 255), y1c = min(max(y1, 0), 255);
      float w00 = wx0 * wy0 * vx0 * vy0;
      float w10 = wx1 * wy0 * vx1 * vy0;
      float w01 = wx0 * wy1 * vx0 * vy1;
      float w11 = wx1 * wy1 * vx1 * vy1;
      int i00 = y0c * 256 + x0c, i10 = y0c * 256 + x1c;
      int i01 = y1c * 256 + x0c, i11 = y1c * 256 + x1c;
      const float* cp = planes + ((size_t)(n * 3 + p) * C_FEAT + cg) * PLANE_HW;
      #pragma unroll 4
      for (int ch = 0; ch < 16; ++ch) {
        const float* c0 = cp + (size_t)ch * PLANE_HW;
        acc[ch] += w00 * c0[i00] + w10 * c0[i10] + w01 * c0[i01] + w11 * c0[i11];
      }
    }
    #pragma unroll
    for (int ch = 0; ch < 16; ++ch)
      x_lds[s_local][cg + ch] = (_Float16)(acc[ch] * (1.0f / 3.0f));
  }
  __syncthreads();

  // ---- phase 1: WMMA MLP ----
  const int lane  = t & 31;
  const int wid   = t >> 5;
  const int row16 = lane & 15;
  const int half  = lane >> 4;   // K-half selector per ISA A-layout

  // A1: x tile, 16x32 f16 (K=32 == C_FEAT)
  v16h a1;
  {
    const _Float16* xr = &x_lds[wid * 16 + row16][0];
    v8h lo = *(const v8h*)(xr + half * 8);        // K =  8*half .. +7
    v8h hi = *(const v8h*)(xr + 16 + half * 8);   // K = 16 + 8*half .. +7
    #pragma unroll
    for (int i = 0; i < 8; ++i) { a1[i] = lo[i]; a1[8 + i] = hi[i]; }
  }

  // GEMM1: h = softplus(x*w1^T + b1), 4 output tiles of 16
  #pragma unroll
  for (int nt = 0; nt < 4; ++nt) {
    v16h b;
    {
      const _Float16* wr = &w1_lds[nt * 16 + row16][0];
      v8h lo = *(const v8h*)(wr + half * 8);
      v8h hi = *(const v8h*)(wr + 16 + half * 8);
      #pragma unroll
      for (int i = 0; i < 8; ++i) { b[i] = lo[i]; b[8 + i] = hi[i]; }
    }
    v8f c = {};
    c = __builtin_amdgcn_wmma_f32_16x16x32_f16(false, a1, false, b,
                                               (short)0, c, false, false);
    const int col = nt * 16 + row16;
    const float bb = b1[col];
    #pragma unroll
    for (int r = 0; r < 8; ++r) {
      int rowm = r + half * 8;   // D layout: VGPR r -> M = r + 8*(lane/16)
      h_lds[wid][rowm][col] = (_Float16)softplus_f(c[r] + bb);
    }
  }
  // per-wave private h_lds region; same-wave LDS ops are in-order -> no barrier

  // A2: h tile 16x64 split into two K=32 fragments
  v16h a2a, a2b;
  {
    const _Float16* hr = &h_lds[wid][row16][0];
    v8h l0 = *(const v8h*)(hr + half * 8);        // K 0..31 half
    v8h h0 = *(const v8h*)(hr + 16 + half * 8);
    v8h l1 = *(const v8h*)(hr + 32 + half * 8);   // K 32..63 half
    v8h h1 = *(const v8h*)(hr + 48 + half * 8);
    #pragma unroll
    for (int i = 0; i < 8; ++i) {
      a2a[i] = l0[i]; a2a[8 + i] = h0[i];
      a2b[i] = l1[i]; a2b[8 + i] = h1[i];
    }
  }

  // GEMM2: out = h*w2^T + b2, 3 output tiles (48 cols, 33 valid)
  #pragma unroll
  for (int ot = 0; ot < 3; ++ot) {
    v16h bA, bB;
    {
      const _Float16* wr = &w2_lds[ot * 16 + row16][0];
      v8h l0 = *(const v8h*)(wr + half * 8);
      v8h h0 = *(const v8h*)(wr + 16 + half * 8);
      v8h l1 = *(const v8h*)(wr + 32 + half * 8);
      v8h h1 = *(const v8h*)(wr + 48 + half * 8);
      #pragma unroll
      for (int i = 0; i < 8; ++i) {
        bA[i] = l0[i]; bA[8 + i] = h0[i];
        bB[i] = l1[i]; bB[8 + i] = h1[i];
      }
    }
    v8f c = {};
    c = __builtin_amdgcn_wmma_f32_16x16x32_f16(false, a2a, false, bA,
                                               (short)0, c, false, false);
    c = __builtin_amdgcn_wmma_f32_16x16x32_f16(false, a2b, false, bB,
                                               (short)0, c, false, false);
    const int o = ot * 16 + row16;   // output channel
    if (o < 33) {
      const float bo = b2[o];
      #pragma unroll
      for (int r = 0; r < 8; ++r) {
        int rowm = r + half * 8;
        int g = block_base + wid * 16 + rowm;
        float v = c[r] + bo;
        if (o == 0) {
          sigma_buf[g] = v;
        } else {
          float rg = sigmoid_f(v) * 1.002f - 0.001f;
          rgb_buf[(size_t)g * 32 + (o - 1)] = (_Float16)rg;
        }
      }
    }
  }
}

// ---------------------------------------------------------------------------
// Kernel 2: alpha compositing. One wave per ray, lane = rgb channel (32 == wave32).
// ---------------------------------------------------------------------------
__global__ __launch_bounds__(256) void composite_kernel(
    const float*    __restrict__ depths,    // (N,R,S,1)
    const _Float16* __restrict__ rgb_buf,   // (TOT,32)
    const float*    __restrict__ sigma_buf, // (TOT)
    float*          __restrict__ out)       // (N,R,32)
{
  const int wid  = threadIdx.x >> 5;
  const int lane = threadIdx.x & 31;
  const int ray  = blockIdx.x * 8 + wid;    // 0 .. N*R-1
  const size_t base = (size_t)ray * SAMP_S;

  float accum = 0.0f, trans = 1.0f;
  float prev_rgb = (float)rgb_buf[base * 32 + lane];
  float prev_sig = sigma_buf[base];
  float prev_d   = depths[base];

  for (int s = 1; s < SAMP_S; ++s) {
    float cur_rgb = (float)rgb_buf[(base + s) * 32 + lane];
    float cur_sig = sigma_buf[base + s];
    float cur_d   = depths[base + s];
    float delta = cur_d - prev_d;
    float cmid  = 0.5f * (prev_rgb + cur_rgb);
    float dens  = softplus_f(0.5f * (prev_sig + cur_sig) - 1.0f);
    float alpha = 1.0f - fast_exp(-delta * dens);
    accum += alpha * trans * cmid;
    trans *= (1.0f - alpha + 1e-10f);
    prev_rgb = cur_rgb; prev_sig = cur_sig; prev_d = cur_d;
  }
  out[(size_t)ray * 32 + lane] = accum * 2.0f - 1.0f;
}

// ---------------------------------------------------------------------------
extern "C" void kernel_launch(void* const* d_in, const int* in_sizes, int n_in,
                              void* d_out, int out_size, void* d_ws, size_t ws_size,
                              hipStream_t stream) {
  const float* planes = (const float*)d_in[0];
  const float* coords = (const float*)d_in[1];
  const float* depths = (const float*)d_in[2];
  const float* w1     = (const float*)d_in[3];
  const float* b1     = (const float*)d_in[4];
  const float* w2     = (const float*)d_in[5];
  const float* b2     = (const float*)d_in[6];

  _Float16* rgb_buf   = (_Float16*)d_ws;                                  // TOT*32 f16 = 50.3 MB
  float*    sigma_buf = (float*)((char*)d_ws + (size_t)TOT * 32 * 2);     // TOT f32   =  3.1 MB

  triplane_mlp_kernel<<<TOT / 128, 256, 0, stream>>>(
      planes, coords, w1, b1, w2, b2, rgb_buf, sigma_buf);
  composite_kernel<<<(NBATCH * RAYS_R) / 8, 256, 0, stream>>>(
      depths, rgb_buf, sigma_buf, (float*)d_out);
}
